// RecActor_23390391894975
// MI455X (gfx1250) — matrix operand order repaired and checked by reading, not compiled, not run on hardware
//
#include <hip/hip_runtime.h>
#include <math.h>

// ---------------------------------------------------------------------------
// CDNA5 (gfx1250) wave32 WMMA types
// ---------------------------------------------------------------------------
typedef __attribute__((ext_vector_type(16))) __bf16 v16bf;
typedef __attribute__((ext_vector_type(8)))  float  v8f;

__device__ __forceinline__ __bf16 f2bf(float f) {
  // round-to-nearest-even f32 -> bf16
  unsigned u = __builtin_bit_cast(unsigned, f);
  u = (u + 0x7FFFu + ((u >> 16) & 1u)) >> 16;
  unsigned short s = (unsigned short)u;
  return __builtin_bit_cast(__bf16, s);
}

// fast transcendentals: keep the serialized GRU scan on single TRANS ops
__device__ __forceinline__ float fast_rcp(float x) { return __builtin_amdgcn_rcpf(x); }
__device__ __forceinline__ float fast_sigmoid(float x) {
  return fast_rcp(1.0f + __expf(-x));
}
__device__ __forceinline__ float fast_tanh(float x) {
#if __has_builtin(__builtin_amdgcn_tanhf)
  return __builtin_amdgcn_tanhf(x);          // gfx1250 v_tanh_f32
#else
  return 1.0f - 2.0f * fast_rcp(1.0f + __expf(2.0f * x));
#endif
}

// ---------------------------------------------------------------------------
// Pack a row-major [K][N] f32 weight matrix into WMMA B-fragment order:
// fragment f = nt*(K/32) + kt holds the 32x16 (KxN) bf16 tile; lane l elem e
// = B[kt*32 + (e>>3)*16 + (l>>4)*8 + (e&7)][nt*16 + (l&15)]  (ISA 7.12.2 B layout).
// Stored contiguously so GEMM lanes do one aligned 32B load per fragment.
// ---------------------------------------------------------------------------
__global__ void __launch_bounds__(256) pack_b_kernel(
    const float* __restrict__ W, __bf16* __restrict__ out, int K, int N) {
  int o = blockIdx.x * 256 + threadIdx.x;      // < K*N
  int f = o >> 9;                              // fragment (512 elems)
  int l = (o >> 4) & 31;                       // lane
  int e = o & 15;                              // element within lane
  int KT = K >> 5;
  int nt = f / KT, kt = f - nt * KT;
  int n = nt * 16 + (l & 15);
  int k = kt * 32 + ((e >> 3) << 4) + (l >> 4) * 8 + (e & 7);
  out[o] = f2bf(W[(size_t)k * N + n]);
}

// ---------------------------------------------------------------------------
// bf16 WMMA GEMM: C[M x N] = op(A[M x K] * Bpack + bias)
// block = 256 threads (8 waves). Block tile = 128 rows x (16*NTILES) cols.
// Each wave keeps NTILES accumulators so every staged A tile / A fragment
// feeds NTILES WMMAs (A global traffic = M*K*(N/(16*NTILES)) reads).
// ---------------------------------------------------------------------------
template <bool A_F32, bool RELU, bool OUT_BF16, int NTILES>
__global__ void __launch_bounds__(256) gemm_wmma_kernel(
    const void* __restrict__ Aptr, const __bf16* __restrict__ Bpack,
    const float* __restrict__ bias, void* __restrict__ Cptr,
    int N, int K, int lda) {
  __shared__ __bf16 As[128][32];
  const int tid = threadIdx.x, wave = tid >> 5, lane = tid & 31;
  const int m0 = blockIdx.x * 128;
  const int nt0 = blockIdx.y * NTILES;
  const int KT = K >> 5;
  const int srow = tid >> 1, shalf = tid & 1;   // each thread stages 16 elems
  v8f acc[NTILES] = {};
  for (int kt = 0; kt < KT; ++kt) {
    if (A_F32) {
      const float* Ap = (const float*)Aptr + (size_t)(m0 + srow) * lda + kt * 32 + shalf * 16;
      #pragma unroll
      for (int j = 0; j < 16; ++j) As[srow][shalf * 16 + j] = f2bf(Ap[j]);
    } else {
      const __bf16* Ap = (const __bf16*)Aptr + (size_t)(m0 + srow) * lda + kt * 32 + shalf * 16;
      #pragma unroll
      for (int j = 0; j < 16; ++j) As[srow][shalf * 16 + j] = Ap[j];
    }
    __syncthreads();
    // A fragment (16x32 bf16, ISA 7.12.2 layout)
    v16bf af;
    const int m = lane & 15, kh = lane >> 4;
    #pragma unroll
    for (int e = 0; e < 16; ++e) {
      int j = e >> 1;
      int kk = ((j >= 4) ? 16 : 0) + kh * 8 + (j & 3) * 2 + (e & 1);
      af[e] = As[wave * 16 + m][kk];
    }
    #pragma unroll
    for (int c = 0; c < NTILES; ++c) {
      v16bf bfrag = *(const v16bf*)(Bpack + (((size_t)(nt0 + c) * KT + kt) << 9) + lane * 16);
      acc[c] = __builtin_amdgcn_wmma_f32_16x16x32_bf16(false, af, false, bfrag,
                                                       (short)0, acc[c], false, false);
    }
    __syncthreads();
  }
  // D layout: lane holds col (lane&15), rows r + 8*(lane>=16)
  const int n = lane & 15, mh = lane >> 4;
  #pragma unroll
  for (int c = 0; c < NTILES; ++c) {
    const int col = (nt0 + c) * 16 + n;
    const float bv = bias[col];
    #pragma unroll
    for (int r = 0; r < 8; ++r) {
      size_t row = (size_t)m0 + wave * 16 + mh * 8 + r;
      float v = acc[c][r] + bv;
      if (RELU) v = fmaxf(v, 0.0f);
      if (OUT_BF16) ((__bf16*)Cptr)[row * N + col] = f2bf(v);
      else          ((float*)Cptr)[row * N + col] = v;
    }
  }
}

// ---------------------------------------------------------------------------
// Persistent GRU scan. Each wave owns 16 batch rows (recurrence is per-row
// independent -> zero inter-wave sync inside a step). Dynamic LDS holds:
//   [0,  64KB)  h state, f32 128x128
//   [64KB,160KB) packed Wh bf16 fragments (96 frags x 1KB), loaded once
// Per step: done-reset, hh = h @ Wh via WMMA (r/z/n col-groups x 4 K-chunks,
// B frags via ds_load_b128), fused sigmoid/tanh (v_rcp/v_tanh) epilogue,
// h update in LDS, ys emitted as bf16 for the head GEMM.
// ---------------------------------------------------------------------------
__global__ void __launch_bounds__(256) gru_scan_kernel(
    const float* __restrict__ h0, const unsigned char* __restrict__ done,
    const float* __restrict__ xi, const __bf16* __restrict__ Whpack,
    const float* __restrict__ bhn, float* __restrict__ h_final,
    __bf16* __restrict__ ys) {
  extern __shared__ char smem_raw[];
  float*  hs  = (float*)smem_raw;                 // 128*128 f32 = 64 KB
  __bf16* whs = (__bf16*)(smem_raw + 65536);      // 128*384 bf16 = 96 KB

  const int tid = threadIdx.x, wave = tid >> 5, lane = tid & 31;
  const int b0 = blockIdx.x * 128;
  const int m = lane & 15, kh = lane >> 4;        // A-frag roles
  const int n = lane & 15, mh = lane >> 4;        // D-frag roles

  #pragma unroll 4
  for (int i = 0; i < 64; ++i) { int p = i * 256 + tid; hs[p] = h0[(size_t)b0 * 128 + p]; }
  #pragma unroll 4
  for (int i = 0; i < 192; ++i) { int p = i * 256 + tid; whs[p] = Whpack[p]; }
  __syncthreads();

  float bhn_r[8];
  #pragma unroll
  for (int c = 0; c < 8; ++c) bhn_r[c] = bhn[c * 16 + n];

  for (int t = 0; t < 256; ++t) {
    // reset carry where done (wave-private rows; flag uniform across lanes)
    const unsigned char* dt = done + (size_t)t * 512 + b0 + wave * 16;
    #pragma unroll
    for (int rl = 0; rl < 16; ++rl) {
      if (dt[rl]) {
        #pragma unroll
        for (int j = 0; j < 4; ++j) hs[(wave * 16 + rl) * 128 + j * 32 + lane] = 0.0f;
      }
    }

    if (t + 1 < 256)
      __builtin_prefetch(xi + (size_t)((t + 1) * 512 + b0 + wave * 16) * 384, 0, 0);

    // cache the wave's 4 A-fragments (16 rows x K=128) for this step
    v16bf af[4];
    #pragma unroll
    for (int kt = 0; kt < 4; ++kt) {
      #pragma unroll
      for (int e = 0; e < 16; ++e) {
        int j = e >> 1;
        int kk = kt * 32 + ((j >= 4) ? 16 : 0) + kh * 8 + (j & 3) * 2 + (e & 1);
        af[kt][e] = f2bf(hs[(wave * 16 + m) * 128 + kk]);
      }
    }

    const float* xit = xi + (size_t)(t * 512 + b0 + wave * 16) * 384;
    for (int c = 0; c < 8; ++c) {
      v8f ar = {}, az = {}, an = {};
      #pragma unroll
      for (int kt = 0; kt < 4; ++kt) {
        v16bf br = *(const v16bf*)(whs + (((size_t)(0 * 8 + c) * 4 + kt) << 9) + lane * 16);
        v16bf bz = *(const v16bf*)(whs + (((size_t)(1 * 8 + c) * 4 + kt) << 9) + lane * 16);
        v16bf bn = *(const v16bf*)(whs + (((size_t)(2 * 8 + c) * 4 + kt) << 9) + lane * 16);
        ar = __builtin_amdgcn_wmma_f32_16x16x32_bf16(false, af[kt], false, br, (short)0, ar, false, false);
        az = __builtin_amdgcn_wmma_f32_16x16x32_bf16(false, af[kt], false, bz, (short)0, az, false, false);
        an = __builtin_amdgcn_wmma_f32_16x16x32_bf16(false, af[kt], false, bn, (short)0, an, false, false);
      }
      #pragma unroll
      for (int r = 0; r < 8; ++r) {
        int ml = mh * 8 + r;                    // row within wave chunk
        int col = c * 16 + n;
        const float* xrow = xit + (size_t)ml * 384;
        float xr = xrow[col], xz = xrow[col + 128], xn = xrow[col + 256];
        float rv = fast_sigmoid(xr + ar[r]);
        float zv = fast_sigmoid(xz + az[r]);
        float nv = fast_tanh(xn + rv * (an[r] + bhn_r[c]));
        float ho = hs[(wave * 16 + ml) * 128 + col];
        float hn2 = (1.0f - zv) * nv + zv * ho;
        hs[(wave * 16 + ml) * 128 + col] = hn2;
        size_t bg = (size_t)(b0 + wave * 16 + ml);
        ys[((size_t)t * 512 + bg) * 128 + col] = f2bf(hn2);
      }
    }
  }
  __syncthreads();
  #pragma unroll 4
  for (int i = 0; i < 64; ++i) { int p = i * 256 + tid; h_final[(size_t)b0 * 128 + p] = hs[p]; }
}

// ---------------------------------------------------------------------------
// Final head GEMM (N=32, 2 col tiles per wave) with bias + action-mask
// epilogue -> masked logits (NEG_INF = -FLT_MAX).
// ---------------------------------------------------------------------------
__global__ void __launch_bounds__(256) gemm_logits_kernel(
    const __bf16* __restrict__ A, const __bf16* __restrict__ Bpack,
    const float* __restrict__ bias, const unsigned char* __restrict__ mask,
    float* __restrict__ out) {
  __shared__ __bf16 As[128][32];
  const int tid = threadIdx.x, wave = tid >> 5, lane = tid & 31;
  const int m0 = blockIdx.x * 128;
  const int srow = tid >> 1, shalf = tid & 1;
  v8f acc[2] = {};
  for (int kt = 0; kt < 4; ++kt) {
    const __bf16* Ap = A + (size_t)(m0 + srow) * 128 + kt * 32 + shalf * 16;
    #pragma unroll
    for (int j = 0; j < 16; ++j) As[srow][shalf * 16 + j] = Ap[j];
    __syncthreads();
    v16bf af;
    const int m = lane & 15, kh = lane >> 4;
    #pragma unroll
    for (int e = 0; e < 16; ++e) {
      int j = e >> 1;
      int kk = ((j >= 4) ? 16 : 0) + kh * 8 + (j & 3) * 2 + (e & 1);
      af[e] = As[wave * 16 + m][kk];
    }
    #pragma unroll
    for (int c = 0; c < 2; ++c) {
      v16bf bfrag = *(const v16bf*)(Bpack + (((size_t)c * 4 + kt) << 9) + lane * 16);
      acc[c] = __builtin_amdgcn_wmma_f32_16x16x32_bf16(false, af, false, bfrag,
                                                       (short)0, acc[c], false, false);
    }
    __syncthreads();
  }
  const int n = lane & 15, mh = lane >> 4;
  #pragma unroll
  for (int c = 0; c < 2; ++c) {
    const int col = c * 16 + n;
    const float bv = bias[col];
    #pragma unroll
    for (int r = 0; r < 8; ++r) {
      size_t row = (size_t)m0 + wave * 16 + mh * 8 + r;
      float v = acc[c][r] + bv;
      out[row * 32 + col] = mask[row * 32 + col] ? v : -3.4028234663852886e+38f;
    }
  }
}

// ---------------------------------------------------------------------------
// Host launch. Inputs (setup_inputs order):
// 0 policy_hidden_state [B,H] f32   1 agents_view [T,B,OBS] f32
// 2 done [T,B] bool                 3 action_mask [T,B,A] bool
// 4 W_enc [512,128]  5 b_enc  6 Wi [128,384]  7 bi  8 Wh [128,384]  9 bhn
// 10 W1 [128,128]   11 b1    12 W2 [128,32]  13 b2
// Output: h_final (65536 f32) ++ masked_logits (4194304 f32)
// ---------------------------------------------------------------------------
extern "C" void kernel_launch(void* const* d_in, const int* in_sizes, int n_in,
                              void* d_out, int out_size, void* d_ws, size_t ws_size,
                              hipStream_t stream) {
  (void)in_sizes; (void)n_in; (void)out_size; (void)ws_size;
  const float* h0   = (const float*)d_in[0];
  const float* obs  = (const float*)d_in[1];
  const unsigned char* done  = (const unsigned char*)d_in[2];
  const unsigned char* amask = (const unsigned char*)d_in[3];
  const float* Wenc = (const float*)d_in[4];
  const float* benc = (const float*)d_in[5];
  const float* Wi   = (const float*)d_in[6];
  const float* bi   = (const float*)d_in[7];
  const float* Wh   = (const float*)d_in[8];
  const float* bhn  = (const float*)d_in[9];
  const float* W1   = (const float*)d_in[10];
  const float* b1   = (const float*)d_in[11];
  const float* W2   = (const float*)d_in[12];
  const float* b2   = (const float*)d_in[13];

  char* ws = (char*)d_ws;
  __bf16* emb   = (__bf16*)(ws);                    // 131072x128 bf16 = 32 MB
  float*  xi    = (float*)(ws + 33554432);          // 131072x384 f32  = 192 MB
  __bf16* ysb   = (__bf16*)(ws + 234881024);        // 131072x128 bf16 = 32 MB
  __bf16* a1    = (__bf16*)(ws + 268435456);        // 131072x128 bf16 = 32 MB
  __bf16* pWenc = (__bf16*)(ws + 301989888);        // 512x128 bf16
  __bf16* pWi   = (__bf16*)(ws + 302120960);        // 128x384 bf16
  __bf16* pWh   = (__bf16*)(ws + 302219264);        // 128x384 bf16
  __bf16* pW1   = (__bf16*)(ws + 302317568);        // 128x128 bf16
  __bf16* pW2   = (__bf16*)(ws + 302350336);        // 128x32  bf16

  // weight pre-swizzle into WMMA fragment layout
  pack_b_kernel<<<dim3(512 * 128 / 256), 256, 0, stream>>>(Wenc, pWenc, 512, 128);
  pack_b_kernel<<<dim3(128 * 384 / 256), 256, 0, stream>>>(Wi,   pWi,   128, 384);
  pack_b_kernel<<<dim3(128 * 384 / 256), 256, 0, stream>>>(Wh,   pWh,   128, 384);
  pack_b_kernel<<<dim3(128 * 128 / 256), 256, 0, stream>>>(W1,   pW1,   128, 128);
  pack_b_kernel<<<dim3(128 * 32  / 256), 256, 0, stream>>>(W2,   pW2,   128, 32);

  // encoder: emb = relu(obs @ W_enc + b_enc)   [M=131072, K=512, N=128]
  gemm_wmma_kernel<true, true, true, 8>
      <<<dim3(1024, 1), 256, 0, stream>>>(obs, pWenc, benc, emb, 128, 512, 512);
  // xi = emb @ Wi + bi                          [M=131072, K=128, N=384]
  gemm_wmma_kernel<false, false, false, 8>
      <<<dim3(1024, 3), 256, 0, stream>>>(emb, pWi, bi, xi, 384, 128, 128);
  // GRU scan (h_final -> d_out[0:65536)); 160KB dynamic LDS (h + Wh frags)
  gru_scan_kernel<<<dim3(4), 256, 163840, stream>>>(h0, done, xi, pWh, bhn,
                                                    (float*)d_out, ysb);
  // head: a1 = relu(ys @ W1 + b1)               [K=128, N=128]
  gemm_wmma_kernel<false, true, true, 8>
      <<<dim3(1024, 1), 256, 0, stream>>>(ysb, pW1, b1, a1, 128, 128, 128);
  // logits = mask ? a1 @ W2 + b2 : -FLT_MAX     [K=128, N=32]
  gemm_logits_kernel<<<dim3(1024), 256, 0, stream>>>(
      a1, pW2, b2, amask, (float*)d_out + 65536);
}
